// LinearAttention_1795296330060
// MI455X (gfx1250) — matrix operand order repaired and checked by reading, not compiled, not run on hardware
//
#include <hip/hip_runtime.h>
#include <hip/hip_bf16.h>

typedef __attribute__((ext_vector_type(16))) __bf16 v16bf;
typedef __attribute__((ext_vector_type(8)))  float  v8f;
typedef __attribute__((ext_vector_type(2)))  __bf16 bf16x2;
typedef __attribute__((ext_vector_type(4)))  __bf16 bf16x4;
typedef __attribute__((ext_vector_type(4)))  int    v4i;

#define BDIM 8
#define NDIM 4096
#define CDIM 1024
#define HDIM 16
#define DH   64

#if defined(__has_builtin)
#if __has_builtin(__builtin_amdgcn_global_load_async_to_lds_b128) && \
    __has_builtin(__builtin_amdgcn_s_wait_asynccnt)
#define HAVE_ASYNC_LDS 1
#endif
#endif

// ---- WMMA fragment loaders (wave32, contiguous k order: K = 16*lh + i in A and B) ----

// A: 16x32 bf16, row-major in LDS with given stride (elements).
// lanes 0-15: M=lane; element i <-> K = 16*lh + i (lanes 0-15 K=0..15, lanes 16-31 K=16..31)
__device__ __forceinline__ v16bf ldsA(const __bf16* s, int stride, int lh, int lnidx) {
  v16bf a;
  const __bf16* row = s + lnidx * stride;
#pragma unroll
  for (int g = 0; g < 8; ++g) {
    int k = 16 * lh + 2 * g;
    a[2 * g]     = row[k];
    a[2 * g + 1] = row[k + 1];
  }
  return a;
}

// B: 32x16 bf16, stored TRANSPOSED in LDS as sT[n][k] (stride elements).
// lanes 0-15 hold K=0..15, lanes 16-31 hold K=16..31; element i <-> K = 16*lh + i
__device__ __forceinline__ v16bf ldsB(const __bf16* sT, int stride, int lh, int lnidx) {
  v16bf b;
  const __bf16* p = sT + lnidx * stride + 16 * lh;
#pragma unroll
  for (int i = 0; i < 16; ++i) b[i] = p[i];
  return b;
}

__device__ __forceinline__ v8f wmma_bf16(v16bf a, v16bf b, v8f c) {
  return __builtin_amdgcn_wmma_f32_16x16x32_bf16(false, a, false, b, (short)0, c, false, false);
}

// ---- Kernel 0: zero the kv state ----
__global__ __launch_bounds__(256) void zero_kernel(float* p, int n) {
  int i = blockIdx.x * 256 + threadIdx.x;
  if (i < n) p[i] = 0.0f;
}

// ---- coalesced tile loaders (f32 global -> bf16 LDS) ----

// X tile: 64 rows x 32 k, pairs along k (contiguous in global), packed b32 LDS stores
__device__ __forceinline__ void load_x_tile(__bf16* xs, const float* xp, int k0, int tid) {
#pragma unroll
  for (int i = 0; i < 4; ++i) {
    int p = tid + i * 256;            // 0..1023 pair index
    int r = p >> 4, kp = p & 15;
    float2 f = *(const float2*)&xp[(long)r * CDIM + k0 + 2 * kp];
    bf16x2 v; v[0] = (__bf16)f.x; v[1] = (__bf16)f.y;
    *(bf16x2*)(xs + r * 32 + 2 * kp) = v;
  }
}

// W tile: 32 k x 64 cols, pairs along col (contiguous in global, coalesced),
// stored transposed in LDS as [col][k]
__device__ __forceinline__ void load_w_tileT(__bf16* wT, const float* w, long rowStride,
                                             int k0, int colBase, int tid) {
#pragma unroll
  for (int i = 0; i < 4; ++i) {
    int p = tid + i * 256;            // 0..1023 pair index
    int kk = p >> 5, cp = p & 31;     // c = 2*cp, 2*cp+1
    float2 f = *(const float2*)&w[(long)(k0 + kk) * rowStride + colBase + 2 * cp];
    wT[(2 * cp) * 32 + kk]     = (__bf16)f.x;
    wT[(2 * cp + 1) * 32 + kk] = (__bf16)f.y;
  }
}

// ---- Kernel 1: k,v tiles -> normalize k -> kv += k^T v (per b,h) ----
__global__ __launch_bounds__(256) void kv_kernel(
    const float* __restrict__ x, const float* __restrict__ qkv_w,
    const float* __restrict__ qkv_b, float* __restrict__ kv_g)
{
  __shared__ __bf16 xs[64 * 32];
  __shared__ __bf16 wkT[64 * 32];   // [col][k]
  __shared__ __bf16 wvT[64 * 32];   // [col][k]
  __shared__ float  kf[64 * 64];    // k rows f32 (pre-norm)
  __shared__ float  norms[64];
  __shared__ __bf16 knT[64 * 64];   // normalized k, transposed: [d][n]
  __shared__ __bf16 vT[64 * 64];    // v transposed: [e][n]

  const int tid   = threadIdx.x;
  const int lane  = tid & 31, wv = tid >> 5;
  const int lh    = lane >> 4, lnidx = lane & 15;
  const int tr    = wv >> 1;           // tile row 0..3
  const int tc0   = (wv & 1) * 2;      // tile cols tc0, tc0+1

  const int chunk = blockIdx.x, h = blockIdx.y, b = blockIdx.z;
  const float* xp = x + ((long)b * NDIM + (long)chunk * 64) * CDIM;
  const int wkCol = CDIM + h * DH;
  const int wvCol = 2 * CDIM + h * DH;

  v8f ak0 = {}, ak1 = {}, av0 = {}, av1 = {};

  for (int k0 = 0; k0 < CDIM; k0 += 32) {
    load_x_tile(xs, xp, k0, tid);
    load_w_tileT(wkT, qkv_w, 3 * CDIM, k0, wkCol, tid);
    load_w_tileT(wvT, qkv_w, 3 * CDIM, k0, wvCol, tid);
    __syncthreads();
    v16bf A   = ldsA(xs + tr * 16 * 32, 32, lh, lnidx);
    v16bf Bk0 = ldsB(wkT + tc0 * 16 * 32, 32, lh, lnidx);
    v16bf Bk1 = ldsB(wkT + (tc0 + 1) * 16 * 32, 32, lh, lnidx);
    v16bf Bv0 = ldsB(wvT + tc0 * 16 * 32, 32, lh, lnidx);
    v16bf Bv1 = ldsB(wvT + (tc0 + 1) * 16 * 32, 32, lh, lnidx);
    ak0 = wmma_bf16(A, Bk0, ak0);
    ak1 = wmma_bf16(A, Bk1, ak1);
    av0 = wmma_bf16(A, Bv0, av0);
    av1 = wmma_bf16(A, Bv1, av1);
    __syncthreads();
  }

  // spill accumulators: k -> f32 (+bias), v -> bf16 transposed (+bias)
#pragma unroll
  for (int e = 0; e < 8; ++e) {
    int r  = tr * 16 + lh * 8 + e;
    int c0 = tc0 * 16 + lnidx, c1 = (tc0 + 1) * 16 + lnidx;
    kf[r * 64 + c0] = ak0[e] + qkv_b[wkCol + c0];
    kf[r * 64 + c1] = ak1[e] + qkv_b[wkCol + c1];
    vT[c0 * 64 + r] = (__bf16)(av0[e] + qkv_b[wvCol + c0]);
    vT[c1 * 64 + r] = (__bf16)(av1[e] + qkv_b[wvCol + c1]);
  }
  __syncthreads();
  if (tid < 64) {                                // L2 norm per k row
    float s = 0.f;
#pragma unroll
    for (int d = 0; d < 64; ++d) { float t = kf[tid * 64 + d]; s += t * t; }
    norms[tid] = 1.0f / fmaxf(sqrtf(s), 1e-12f);
  }
  __syncthreads();
#pragma unroll
  for (int i = 0; i < 16; ++i) {                 // normalized k, transposed [d][n]
    int idx = tid + i * 256;
    int n = idx >> 6, d = idx & 63;
    knT[d * 64 + n] = (__bf16)(kf[n * 64 + d] * norms[n]);
  }
  __syncthreads();

  // GEMM2: ktv[d][e] = sum_n knT[d][n] * v[n][e]  (M=d, N=e, K=n=64)
  v8f c0acc = {}, c1acc = {};
#pragma unroll
  for (int k0 = 0; k0 < 64; k0 += 32) {
    v16bf A  = ldsA(knT + tr * 16 * 64 + k0, 64, lh, lnidx);
    v16bf B0 = ldsB(vT + tc0 * 16 * 64 + k0, 64, lh, lnidx);
    v16bf B1 = ldsB(vT + (tc0 + 1) * 16 * 64 + k0, 64, lh, lnidx);
    c0acc = wmma_bf16(A, B0, c0acc);
    c1acc = wmma_bf16(A, B1, c1acc);
  }
  float* kvp = kv_g + (long)(b * HDIM + h) * (DH * DH);
#pragma unroll
  for (int e = 0; e < 8; ++e) {
    int d = tr * 16 + lh * 8 + e;
    atomicAdd(&kvp[d * 64 + tc0 * 16 + lnidx], c0acc[e]);
    atomicAdd(&kvp[d * 64 + (tc0 + 1) * 16 + lnidx], c1acc[e]);
  }
}

// ---- Kernel 2: q tiles -> normalize -> q @ kv -> attn (bf16) ----
__global__ __launch_bounds__(256) void q_attn_kernel(
    const float* __restrict__ x, const float* __restrict__ qkv_w,
    const float* __restrict__ qkv_b, const float* __restrict__ kv_g,
    __bf16* __restrict__ attn)
{
  __shared__ __bf16 xs[64 * 32];
  __shared__ __bf16 wqT[64 * 32];
  __shared__ float  qf[64 * 64];
  __shared__ float  norms[64];
  __shared__ __bf16 qn[64 * 64];    // [n][d] (natural A layout)
  __shared__ __bf16 kvT[64 * 64];   // [e][d] (B transposed)

  const int tid  = threadIdx.x;
  const int lane = tid & 31, wv = tid >> 5;
  const int lh   = lane >> 4, lnidx = lane & 15;
  const int tr   = wv >> 1, tc0 = (wv & 1) * 2;

  const int chunk = blockIdx.x, h = blockIdx.y, b = blockIdx.z;
  const float* xp = x + ((long)b * NDIM + (long)chunk * 64) * CDIM;
  const int wqCol = h * DH;
  const float* kvp = kv_g + (long)(b * HDIM + h) * (DH * DH);

  // stage kv state (transposed, bf16)
#pragma unroll
  for (int i = 0; i < 16; ++i) {
    int idx = tid + i * 256;
    int d = idx >> 6, e = idx & 63;
    kvT[e * 64 + d] = (__bf16)kvp[d * 64 + e];
  }

  v8f a0 = {}, a1 = {};
  for (int k0 = 0; k0 < CDIM; k0 += 32) {
    load_x_tile(xs, xp, k0, tid);
    load_w_tileT(wqT, qkv_w, 3 * CDIM, k0, wqCol, tid);
    __syncthreads();
    v16bf A  = ldsA(xs + tr * 16 * 32, 32, lh, lnidx);
    v16bf B0 = ldsB(wqT + tc0 * 16 * 32, 32, lh, lnidx);
    v16bf B1 = ldsB(wqT + (tc0 + 1) * 16 * 32, 32, lh, lnidx);
    a0 = wmma_bf16(A, B0, a0);
    a1 = wmma_bf16(A, B1, a1);
    __syncthreads();
  }

#pragma unroll
  for (int e = 0; e < 8; ++e) {
    int r  = tr * 16 + lh * 8 + e;
    int c0 = tc0 * 16 + lnidx, c1 = (tc0 + 1) * 16 + lnidx;
    qf[r * 64 + c0] = a0[e] + qkv_b[wqCol + c0];
    qf[r * 64 + c1] = a1[e] + qkv_b[wqCol + c1];
  }
  __syncthreads();
  if (tid < 64) {
    float s = 0.f;
#pragma unroll
    for (int d = 0; d < 64; ++d) { float t = qf[tid * 64 + d]; s += t * t; }
    norms[tid] = 1.0f / fmaxf(sqrtf(s), 1e-12f);
  }
  __syncthreads();
#pragma unroll
  for (int i = 0; i < 16; ++i) {
    int idx = tid + i * 256;
    int n = idx >> 6, d = idx & 63;
    qn[n * 64 + d] = (__bf16)(qf[n * 64 + d] * norms[n]);
  }
  __syncthreads();

  // GEMM2: out[n][e] = sum_d qn[n][d] * kv[d][e]
  v8f c0acc = {}, c1acc = {};
#pragma unroll
  for (int k0 = 0; k0 < 64; k0 += 32) {
    v16bf A  = ldsA(qn + tr * 16 * 64 + k0, 64, lh, lnidx);
    v16bf B0 = ldsB(kvT + tc0 * 16 * 64 + k0, 64, lh, lnidx);
    v16bf B1 = ldsB(kvT + (tc0 + 1) * 16 * 64 + k0, 64, lh, lnidx);
    c0acc = wmma_bf16(A, B0, c0acc);
    c1acc = wmma_bf16(A, B1, c1acc);
  }
  __bf16* op = attn + ((long)b * NDIM + (long)chunk * 64) * CDIM + h * DH;
#pragma unroll
  for (int e = 0; e < 8; ++e) {
    int r = tr * 16 + lh * 8 + e;
    op[(long)r * CDIM + tc0 * 16 + lnidx]       = (__bf16)c0acc[e];
    op[(long)r * CDIM + (tc0 + 1) * 16 + lnidx] = (__bf16)c1acc[e];
  }
}

// ---- Kernel 3: out = attn @ proj_w + proj_b (f32 out) ----
__global__ __launch_bounds__(256) void proj_kernel(
    const __bf16* __restrict__ attn, const float* __restrict__ proj_w,
    const float* __restrict__ proj_b, float* __restrict__ out)
{
  __shared__ __bf16 as_[64 * 32];
  __shared__ __bf16 wT[64 * 32];

  const int tid  = threadIdx.x;
  const int lane = tid & 31, wv = tid >> 5;
  const int lh   = lane >> 4, lnidx = lane & 15;
  const int tr   = wv >> 1, tc0 = (wv & 1) * 2;

  const int colBase = blockIdx.x * 64;
  const long rowBase = (long)blockIdx.y * 64;

  v8f a0 = {}, a1 = {};
  for (int k0 = 0; k0 < CDIM; k0 += 32) {
    // A tile: already bf16 in global -> pure byte copy into LDS.
#if defined(HAVE_ASYNC_LDS)
    {
      // 64 rows x 32 bf16 = 4KB: one b128 per thread, ASYNCcnt-tracked DMA
      int r = tid >> 2, kq = tid & 3;   // 4 x 16B chunks per row
      __bf16* gsrc = const_cast<__bf16*>(&attn[(rowBase + r) * CDIM + k0 + 8 * kq]);
      __builtin_amdgcn_global_load_async_to_lds_b128(
          (__attribute__((address_space(1))) v4i*)(void*)gsrc,
          (__attribute__((address_space(3))) v4i*)(void*)&as_[r * 32 + 8 * kq],
          0, 0);
    }
#else
#pragma unroll
    for (int i = 0; i < 2; ++i) {
      int q = tid + i * 256;            // 0..511 quad index
      int r = q >> 3, kq = q & 7;
      bf16x4 v = *(const bf16x4*)&attn[(rowBase + r) * CDIM + k0 + 4 * kq];
      *(bf16x4*)(as_ + r * 32 + 4 * kq) = v;
    }
#endif
    load_w_tileT(wT, proj_w, CDIM, k0, colBase, tid);
#if defined(HAVE_ASYNC_LDS)
    __builtin_amdgcn_s_wait_asynccnt(0);
#endif
    __syncthreads();
    v16bf A  = ldsA(as_ + tr * 16 * 32, 32, lh, lnidx);
    v16bf B0 = ldsB(wT + tc0 * 16 * 32, 32, lh, lnidx);
    v16bf B1 = ldsB(wT + (tc0 + 1) * 16 * 32, 32, lh, lnidx);
    a0 = wmma_bf16(A, B0, a0);
    a1 = wmma_bf16(A, B1, a1);
    __syncthreads();
  }
#pragma unroll
  for (int e = 0; e < 8; ++e) {
    int r  = tr * 16 + lh * 8 + e;
    int c0 = colBase + tc0 * 16 + lnidx;
    int c1 = colBase + (tc0 + 1) * 16 + lnidx;
    out[(rowBase + r) * CDIM + c0] = a0[e] + proj_b[c0];
    out[(rowBase + r) * CDIM + c1] = a1[e] + proj_b[c1];
  }
}

extern "C" void kernel_launch(void* const* d_in, const int* in_sizes, int n_in,
                              void* d_out, int out_size, void* d_ws, size_t ws_size,
                              hipStream_t stream) {
  const float* x      = (const float*)d_in[0];
  const float* qkv_w  = (const float*)d_in[1];
  const float* qkv_b  = (const float*)d_in[2];
  const float* proj_w = (const float*)d_in[3];
  const float* proj_b = (const float*)d_in[4];
  float* out = (float*)d_out;

  const size_t kv_elems = (size_t)BDIM * HDIM * DH * DH;            // 512K f32 = 2MB
  float*  kv_g = (float*)d_ws;
  __bf16* attn = (__bf16*)((char*)d_ws + kv_elems * sizeof(float)); // 64MB bf16

  zero_kernel<<<(int)(kv_elems / 256), 256, 0, stream>>>(kv_g, (int)kv_elems);

  dim3 g1(NDIM / 64, HDIM, BDIM);
  kv_kernel<<<g1, 256, 0, stream>>>(x, qkv_w, qkv_b, kv_g);
  q_attn_kernel<<<g1, 256, 0, stream>>>(x, qkv_w, qkv_b, kv_g, attn);

  dim3 g3(CDIM / 64, (BDIM * NDIM) / 64, 1);
  proj_kernel<<<g3, 256, 0, stream>>>(attn, proj_w, proj_b, out);
}
